// RNN_1194000908562
// MI455X (gfx1250) — matrix operand order repaired and checked by reading, not compiled, not run on hardware
//
#include <hip/hip_runtime.h>
#include <hip/hip_bf16.h>

// CDNA5 (gfx1250) wave32 WMMA RNN.
// Roofline: 2.8 GFLOP vs 27 MB -> memory bound (~1.2us) only with matrix cores.
// Split-f16 WMMA (hi*hi + hi*lo + lo*hi) gives ~f32 accuracy at f16 rate.

typedef __attribute__((ext_vector_type(16))) _Float16 v16h;
typedef __attribute__((ext_vector_type(8)))  float    v8f;

#if __has_builtin(__builtin_amdgcn_tanhf)
#define TANHF(x) __builtin_amdgcn_tanhf(x)   // gfx1250 V_TANH_F32 (TRANS op)
#else
#define TANHF(x) tanhf(x)
#endif

#define T_STEPS 10
#define HID 32
#define ROW_STRIDE 34                         // padded LDS row stride (floats)
#define WAVE_LDS (16 * ROW_STRIDE + 16 * T_STEPS)  // 544 h-tile + 160 x = 704 floats

__global__ __launch_bounds__(256) void rnn_wmma_kernel(
    const float* __restrict__ x,      // [B, T, 1]
    const float* __restrict__ h0g,    // [1, B, H]
    const float* __restrict__ Wih,    // [H, 1]
    const float* __restrict__ Whh,    // [H, H]
    const float* __restrict__ bih,    // [H]
    const float* __restrict__ bhh,    // [H]
    const float* __restrict__ Wout,   // [1, H]
    const float* __restrict__ boutp,  // [1]
    float* __restrict__ outs,         // [B, T]
    float* __restrict__ hT)           // [B, H]
{
  __shared__ float smem[8 * WAVE_LDS];
  const int lane = threadIdx.x & 31;
  const int wid  = threadIdx.x >> 5;
  const int half = lane >> 4;          // lane-half (0/1)
  const int n16  = lane & 15;
  const long b0  = ((long)blockIdx.x * 8 + wid) * 16;   // first batch row of this wave

  float* ldsH = smem + wid * WAVE_LDS;        // 16 x 32 h tile (stride 34)
  float* ldsX = ldsH + 16 * ROW_STRIDE;       // 16 x 10 x tile

  // ---- per-lane constants (hidden index n16 / n16+16)
  const float wihA  = Wih[n16],           wihB  = Wih[n16 + 16];
  const float biasA = bih[n16] + bhh[n16];
  const float biasB = bih[n16 + 16] + bhh[n16 + 16];
  const float woutA = Wout[n16],          woutB = Wout[n16 + 16];
  const float boutv = boutp[0];

  // ---- stage this wave's x rows (16 rows x 10 steps = 160 floats) into LDS
  {
    const float* xg = x + b0 * T_STEPS;
#pragma unroll
    for (int i = 0; i < 5; ++i)
      ldsX[lane + 32 * i] = xg[lane + 32 * i];
  }

  // ---- build split-f16 B fragments of Whh^T once.
  // B (16-bit, K=32 x N=16): VGPR v, lanes 0-15 hold K={2v,2v+1}, lanes 16-31
  // hold K={16+2v,16+2v+1}; N = lane%16 (+16 for the second column tile).
  // B[k][n] = Whh[n][k] (row-major Whh), pairs are contiguous in a Whh row.
  v16h bh0, bl0, bh1, bl1;
#pragma unroll
  for (int v = 0; v < 8; ++v) {
    const int k0 = 2 * v + 16 * half;
    {
      const float2 w = *(const float2*)(Whh + n16 * HID + k0);
      _Float16 hx = (_Float16)w.x, hy = (_Float16)w.y;
      bh0[2 * v] = hx;  bh0[2 * v + 1] = hy;
      bl0[2 * v]     = (_Float16)(w.x - (float)hx);
      bl0[2 * v + 1] = (_Float16)(w.y - (float)hy);
    }
    {
      const float2 w = *(const float2*)(Whh + (n16 + 16) * HID + k0);
      _Float16 hx = (_Float16)w.x, hy = (_Float16)w.y;
      bh1[2 * v] = hx;  bh1[2 * v + 1] = hy;
      bl1[2 * v]     = (_Float16)(w.x - (float)hx);
      bl1[2 * v + 1] = (_Float16)(w.y - (float)hy);
    }
  }

  // ---- initial hidden state -> LDS tile (C/D layout row M = v + 8*half)
#pragma unroll
  for (int v = 0; v < 8; ++v) {
    const int rowM = v + 8 * half;
    ldsH[rowM * ROW_STRIDE + n16]      = h0g[(b0 + rowM) * HID + n16];
    ldsH[rowM * ROW_STRIDE + 16 + n16] = h0g[(b0 + rowM) * HID + 16 + n16];
  }

  v8f acc0, acc1;
  float h0a[8], h1a[8];

  for (int t = 0; t < T_STEPS; ++t) {
    __syncthreads();   // order: previous h stores -> this step's A loads

    // ---- A fragments (split f16) from LDS h tile.
    // A (16-bit, M=16 x K=32): lane: M=lane%16; VGPR v: K={2v,2v+1}+16*half
    // (same k order as the B fragments).
    v16h ah, al;
#pragma unroll
    for (int v = 0; v < 8; ++v) {
      const int k0 = 2 * v + 16 * half;
      const float2 f = *(const float2*)(ldsH + n16 * ROW_STRIDE + k0);
      _Float16 hx = (_Float16)f.x, hy = (_Float16)f.y;
      ah[2 * v] = hx;  ah[2 * v + 1] = hy;
      al[2 * v]     = (_Float16)(f.x - (float)hx);
      al[2 * v + 1] = (_Float16)(f.y - (float)hy);
    }

    // ---- C init: x_t * Wih + (b_ih + b_hh), C/D layout row M = v + 8*half
#pragma unroll
    for (int v = 0; v < 8; ++v) {
      const int rowM = v + 8 * half;
      const float xr = ldsX[rowM * T_STEPS + t];
      acc0[v] = fmaf(xr, wihA, biasA);
      acc1[v] = fmaf(xr, wihB, biasB);
    }

    // ---- 3-term split-f16 GEMM: ~f32 accuracy, f16 matrix-core rate
    acc0 = __builtin_amdgcn_wmma_f32_16x16x32_f16(false, ah, false, bh0, (short)0, acc0, false, false);
    acc1 = __builtin_amdgcn_wmma_f32_16x16x32_f16(false, ah, false, bh1, (short)0, acc1, false, false);
    acc0 = __builtin_amdgcn_wmma_f32_16x16x32_f16(false, ah, false, bl0, (short)0, acc0, false, false);
    acc1 = __builtin_amdgcn_wmma_f32_16x16x32_f16(false, ah, false, bl1, (short)0, acc1, false, false);
    acc0 = __builtin_amdgcn_wmma_f32_16x16x32_f16(false, al, false, bh0, (short)0, acc0, false, false);
    acc1 = __builtin_amdgcn_wmma_f32_16x16x32_f16(false, al, false, bh1, (short)0, acc1, false, false);

    __syncthreads();   // A loads done before h tile is overwritten

    // ---- tanh, write h back to LDS, and output projection for this step
#pragma unroll
    for (int v = 0; v < 8; ++v) {
      const int rowM = v + 8 * half;
      const float hv0 = TANHF(acc0[v]);
      const float hv1 = TANHF(acc1[v]);
      h0a[v] = hv0;  h1a[v] = hv1;
      ldsH[rowM * ROW_STRIDE + n16]      = hv0;
      ldsH[rowM * ROW_STRIDE + 16 + n16] = hv1;

      // out[b,t] = sum_n h[n]*Wout[n] + b_out : 16-lane xor-tree reduction
      float s = fmaf(hv0, woutA, hv1 * woutB);
      s += __shfl_xor(s, 1, 16);
      s += __shfl_xor(s, 2, 16);
      s += __shfl_xor(s, 4, 16);
      s += __shfl_xor(s, 8, 16);
      if (n16 == 0)
        outs[(b0 + rowM) * T_STEPS + t] = s + boutv;
    }
  }

  // ---- final hidden state h_T -> global
#pragma unroll
  for (int v = 0; v < 8; ++v) {
    const int rowM = v + 8 * half;
    hT[(b0 + rowM) * HID + n16]      = h0a[v];
    hT[(b0 + rowM) * HID + 16 + n16] = h1a[v];
  }
}

extern "C" void kernel_launch(void* const* d_in, const int* in_sizes, int n_in,
                              void* d_out, int out_size, void* d_ws, size_t ws_size,
                              hipStream_t stream) {
  const float* x    = (const float*)d_in[0];
  const float* h0   = (const float*)d_in[1];
  const float* Wih  = (const float*)d_in[2];
  const float* Whh  = (const float*)d_in[3];
  const float* bih  = (const float*)d_in[4];
  const float* bhh  = (const float*)d_in[5];
  const float* Wout = (const float*)d_in[6];
  const float* bout = (const float*)d_in[7];

  const int B = in_sizes[0] / T_STEPS;     // x has B*T*1 elements
  float* outs = (float*)d_out;             // [B,T] then [B,H] concatenated
  float* hT   = outs + (size_t)B * T_STEPS;

  const int waves  = B / 16;               // one wave per 16 batch rows
  const int blocks = waves / 8;            // 8 waves (256 threads) per block
  rnn_wmma_kernel<<<blocks, 256, 0, stream>>>(x, h0, Wih, Whh, bih, bhh,
                                              Wout, bout, outs, hT);
}
